// MultiActorCriticRNN_16810501996979
// MI455X (gfx1250) — matrix operand order repaired and checked by reading, not compiled, not run on hardware
//
#include <hip/hip_runtime.h>
#include <hip/hip_bf16.h>
#include <stdint.h>

typedef __bf16 bf16;
typedef __attribute__((ext_vector_type(16))) bf16  v16bf;
typedef __attribute__((ext_vector_type(8)))  float v8f;

#define A_   4
#define T_   128
#define B_   256
#define H_   128
#define ACT_ 8
#define G3_  384            // 3*H
#define M_   (T_ * B_)      // 32768 rows per agent

#define USE_ASYNC_XI 1      // async double-buffer of xi in the GRU scan

struct alignas(16) bh8 { bf16 h[8]; };

// ---------------- fast transcendentals (CDNA5 has V_TANH_F32) ----------------
#if defined(__has_builtin)
#if __has_builtin(__builtin_amdgcn_tanhf)
#define FAST_TANH(x) __builtin_amdgcn_tanhf(x)
#endif
#endif
#ifndef FAST_TANH
#define FAST_TANH(x) tanhf(x)
#endif
static __device__ inline float fast_sigmoid(float x) {
  return __builtin_fmaf(0.5f, FAST_TANH(0.5f * x), 0.5f);
}

// ---------------- WMMA helpers (CDNA5 wave32 layouts, ISA 7.12.2) ----------------

static __device__ inline v8f wmma_bf16(v16bf a, v16bf b, v8f c) {
  return __builtin_amdgcn_wmma_f32_16x16x32_bf16(
      false, a, false, b, (short)0, c, false, false);
}

// A fragment: 16x32 bf16 tile from row-major LDS (leading dim ld, 16B-aligned rows).
// Two contiguous 16B groups per lane -> exactly 2 x ds_load_b128.
static __device__ inline v16bf load_a_frag(const bf16* src, int lane, int r, int kk, int ld) {
  const int row = r * 16 + (lane & 15);
  const int hi  = lane >> 4;
  const bf16* p = src + row * ld + kk * 32 + hi * 8;
  union { v16bf v; uint4 q[2]; } u;
  u.q[0] = *(const uint4*)p;          // elements 0..7   (K = kk*32 + hi*8 + 0..7)
  u.q[1] = *(const uint4*)(p + 16);   // elements 8..15  (K = kk*32 + 16 + hi*8 + 0..7)
  return u.v;
}

// B fragment from pre-packed fragment-major weights: one aligned 32B load.
static __device__ inline v16bf load_b_packed(const v16bf* base, int lane, int c, int kk, int nkk) {
  return base[(size_t)((c * nkk + kk) * 32 + lane)];
}

#if USE_ASYNC_XI
// async copy 16B global -> LDS; tracked by ASYNCcnt
static __device__ inline void async_ld16(uint32_t lds_off, const void* gaddr) {
  asm volatile("global_load_async_to_lds_b128 %0, %1, off"
               :: "v"(lds_off), "v"(gaddr) : "memory");
}
#endif

// ---------------- Kernel 0: pack weights into B-fragment-major bf16 ----------------
// dst layout per agent: [c][kk][lane][i]; element: K = kk*32 + (lane>>4)*8 + (i&7) + (i>>3)*16,
// N = c*16 + (lane&15). Cols >= Nsrc are zero-padded.
__global__ void k_pack_b(const float* __restrict__ src, bf16* __restrict__ dst,
                         int K, int N, int Nsrc) {
  int i = blockIdx.x * 256 + threadIdx.x;
  int per_agent = K * N;
  if (i >= A_ * per_agent) return;
  int a    = i / per_agent;
  int rem  = i - a * per_agent;
  int ie   = rem & 15;
  int lane = (rem >> 4) & 31;
  int fidx = rem >> 9;          // c*nkk + kk
  int nkk  = K >> 5;
  int kk   = fidx % nkk;
  int c    = fidx / nkk;
  int col  = c * 16 + (lane & 15);
  int krow = kk * 32 + (lane >> 4) * 8 + (ie & 7) + ((ie >> 3) << 4);
  float v = (col < Nsrc) ? src[((size_t)a * K + krow) * Nsrc + col] : 0.0f;
  dst[i] = (bf16)v;
}

__global__ void k_std(const float* __restrict__ log_std, float* __restrict__ std_out) {
  int i = threadIdx.x;
  if (i < A_ * ACT_) std_out[i] = __expf(log_std[i]);
}

// ---------------- Kernel 1: xi = tanh(obs@We + be) @ Wi + bi (stored bf16) ----------------
__global__ __launch_bounds__(256) void k_emb_xi(
    const float* __restrict__ obs, const float* __restrict__ be,
    const float* __restrict__ bi,
    const bf16* __restrict__ We_p, const bf16* __restrict__ Wi_p,
    bf16* __restrict__ xi_out) {
  __shared__ alignas(32) bf16 s_in[32 * H_];
  __shared__ alignas(32) bf16 s_emb[32 * H_];
  const int a = blockIdx.x;
  const int m0 = blockIdx.y * 32;
  const int tid = threadIdx.x;
  const int lane = tid & 31, wave = tid >> 5;

  // stage obs tile as bf16: 16 contiguous elems (32B) per thread
  const float* obs_a = obs + ((size_t)a * M_ + m0) * H_;
  {
    const float4* src4 = (const float4*)obs_a;
    v16bf v;
#pragma unroll
    for (int q = 0; q < 4; ++q) {
      float4 f = src4[tid * 4 + q];
      v[q * 4 + 0] = (bf16)f.x; v[q * 4 + 1] = (bf16)f.y;
      v[q * 4 + 2] = (bf16)f.z; v[q * 4 + 3] = (bf16)f.w;
    }
    ((v16bf*)s_in)[tid] = v;
  }
  __syncthreads();

  const v16bf* Wep = (const v16bf*)We_p + (size_t)a * (H_ * H_ / 16);
  v8f acc0 = {}, acc1 = {};
#pragma unroll
  for (int kk = 0; kk < 4; ++kk) {
    v16bf bw  = load_b_packed(Wep, lane, wave, kk, 4);
    v16bf af0 = load_a_frag(s_in, lane, 0, kk, H_);
    v16bf af1 = load_a_frag(s_in, lane, 1, kk, H_);
    acc0 = wmma_bf16(af0, bw, acc0);
    acc1 = wmma_bf16(af1, bw, acc1);
  }
  {
    const int col = wave * 16 + (lane & 15);
    const float bev = be[a * H_ + col];
    const int rb = (lane >> 4) * 8;
#pragma unroll
    for (int v = 0; v < 8; ++v) {
      s_emb[(rb + v) * H_ + col]      = (bf16)FAST_TANH(acc0[v] + bev);
      s_emb[(16 + rb + v) * H_ + col] = (bf16)FAST_TANH(acc1[v] + bev);
    }
  }
  __syncthreads();

  const v16bf* Wip = (const v16bf*)Wi_p + (size_t)a * (H_ * G3_ / 16);
  bf16* xi_a = xi_out + ((size_t)a * M_ + m0) * G3_;
  v8f acc[2][3];
#pragma unroll
  for (int r = 0; r < 2; ++r)
#pragma unroll
    for (int j = 0; j < 3; ++j) { v8f z = {}; acc[r][j] = z; }
#pragma unroll
  for (int kk = 0; kk < 4; ++kk) {
    v16bf af0 = load_a_frag(s_emb, lane, 0, kk, H_);
    v16bf af1 = load_a_frag(s_emb, lane, 1, kk, H_);
#pragma unroll
    for (int j = 0; j < 3; ++j) {
      v16bf bw = load_b_packed(Wip, lane, wave * 3 + j, kk, 4);
      acc[0][j] = wmma_bf16(af0, bw, acc[0][j]);
      acc[1][j] = wmma_bf16(af1, bw, acc[1][j]);
    }
  }
#pragma unroll
  for (int j = 0; j < 3; ++j) {
    const int cc = (wave * 3 + j) * 16 + (lane & 15);
    const float biv = bi[a * G3_ + cc];
    const int rb = (lane >> 4) * 8;
#pragma unroll
    for (int v = 0; v < 8; ++v) {
      xi_a[(size_t)(rb + v) * G3_ + cc]      = (bf16)(acc[0][j][v] + biv);
      xi_a[(size_t)(16 + rb + v) * G3_ + cc] = (bf16)(acc[1][j][v] + biv);
    }
  }
}

// ---------------- Kernel 2: GRU scan ----------------
// grid (A, B/16). Wh fragments register-resident for all T; xi for step t+1
// async-prefetched into an LDS ping-pong buffer while step t's WMMAs run.
__global__ __launch_bounds__(256) void k_gru_scan(
    const float* __restrict__ hstate, const bf16* __restrict__ xi,
    const float* __restrict__ bhn, const unsigned char* __restrict__ done,
    const bf16* __restrict__ Wh_p, bf16* __restrict__ ys_bf,
    float* __restrict__ hT_out) {
  __shared__ float s_h[16 * H_];
  __shared__ alignas(32) bf16 s_hbf[16 * H_];
  __shared__ float s_hh[16 * G3_];
  __shared__ alignas(16) bf16 s_xi[2][16 * G3_];
  __shared__ float s_bhn[H_];
  const int a = blockIdx.x;
  const int b0 = blockIdx.y * 16;
  const int tid = threadIdx.x;
  const int lane = tid & 31, wave = tid >> 5;
  const int rrow = tid >> 4;
  const int c0 = (tid & 15) * 8;
  const int hbase = rrow * H_ + c0;

  const v16bf* Whp = (const v16bf*)Wh_p + (size_t)a * (H_ * G3_ / 16);
  v16bf bfr[3][4];
#pragma unroll
  for (int j = 0; j < 3; ++j)
#pragma unroll
    for (int kk = 0; kk < 4; ++kk)
      bfr[j][kk] = load_b_packed(Whp, lane, wave * 3 + j, kk, 4);

  const float* h0 = hstate + ((size_t)a * B_ + b0) * H_;
#pragma unroll
  for (int u = 0; u < 8; ++u) s_h[hbase + u] = h0[hbase + u];
  if (tid < H_) s_bhn[tid] = bhn[a * H_ + tid];

  const bf16* xi_a = xi + ((size_t)a * M_ + b0) * G3_;
  const unsigned char* done_a = done + (size_t)a * M_ + b0;
  bf16* ys_a = ys_bf + ((size_t)a * M_ + b0) * H_;

#if USE_ASYNC_XI
#pragma unroll
  for (int q = 0; q < 3; ++q) {
    int ch = q * 256 + tid;
    async_ld16((uint32_t)(uintptr_t)(&s_xi[0][ch * 8]), xi_a + (size_t)ch * 8);
  }
#endif
  __syncthreads();

  int cur = 0;
  for (int t = 0; t < T_; ++t) {
    const unsigned char* dt = done_a + (size_t)t * B_;
    // (a) reset-on-done + bf16 shadow
    {
      const bool dflag = dt[rrow] != 0;
#pragma unroll
      for (int u = 0; u < 8; ++u) {
        float hv = s_h[hbase + u];
        if (dflag) hv = 0.0f;
        s_h[hbase + u] = hv;
        s_hbf[hbase + u] = (bf16)hv;
      }
    }
    __syncthreads();   // all waves past previous step's reads; s_hbf ready

#if USE_ASYNC_XI
    {
      const int tn = (t + 1 < T_) ? t + 1 : t;
      const bf16* xin = xi_a + (size_t)tn * B_ * G3_;
#pragma unroll
      for (int q = 0; q < 3; ++q) {
        int ch = q * 256 + tid;
        async_ld16((uint32_t)(uintptr_t)(&s_xi[cur ^ 1][ch * 8]), xin + (size_t)ch * 8);
      }
    }
#else
    {
      const bf16* xin = xi_a + (size_t)t * B_ * G3_;
      const uint4* s4 = (const uint4*)xin;
      uint4* d4 = (uint4*)s_xi[cur];
#pragma unroll
      for (int q = 0; q < 3; ++q) d4[q * 256 + tid] = s4[q * 256 + tid];
    }
#endif

    // (b,c) hh = h @ Wh : [16,384], wave owns 3 col-tiles
    {
      v16bf af[4];
#pragma unroll
      for (int kk = 0; kk < 4; ++kk) af[kk] = load_a_frag(s_hbf, lane, 0, kk, H_);
#pragma unroll
      for (int j = 0; j < 3; ++j) {
        v8f acc = {};
#pragma unroll
        for (int kk = 0; kk < 4; ++kk) acc = wmma_bf16(af[kk], bfr[j][kk], acc);
        const int col = (wave * 3 + j) * 16 + (lane & 15);
        const int rb  = (lane >> 4) * 8;
#pragma unroll
        for (int v = 0; v < 8; ++v) s_hh[(rb + v) * G3_ + col] = acc[v];
      }
    }
#if USE_ASYNC_XI
    asm volatile("s_wait_asynccnt 0x3" ::: "memory");   // current step's xi landed
#endif
    __syncthreads();   // s_hh + s_xi[cur] visible to all waves

    // (d) gates + state update (r|z|n, bias only on n-hidden)
    {
      const bf16* xrow = &s_xi[cur][rrow * G3_];
      const float* hhrow = &s_hh[rrow * G3_];
      float hnew[8];
#pragma unroll
      for (int u = 0; u < 8; ++u) {
        const int col = c0 + u;
        float xr = (float)xrow[col];
        float xz = (float)xrow[H_ + col];
        float xn = (float)xrow[2 * H_ + col];
        float rg = fast_sigmoid(xr + hhrow[col]);
        float zg = fast_sigmoid(xz + hhrow[H_ + col]);
        float ng = FAST_TANH(xn + rg * (hhrow[2 * H_ + col] + s_bhn[col]));
        float hv = s_h[hbase + u];
        hnew[u] = (1.0f - zg) * ng + zg * hv;
        s_h[hbase + u] = hnew[u];
      }
      bh8 yv;
#pragma unroll
      for (int u = 0; u < 8; ++u) yv.h[u] = (bf16)hnew[u];
      *(bh8*)(ys_a + (size_t)t * B_ * H_ + hbase) = yv;
      if (t == T_ - 1) {
        float* hp = hT_out + ((size_t)a * B_ + b0 + rrow) * H_ + c0;
#pragma unroll
        for (int u = 0; u < 8; ++u) hp[u] = hnew[u];
      }
    }
    cur ^= 1;
  }
}

// ---------------- Kernel 3: actor/critic heads ----------------
// a1 = tanh(ys@Wa1+ba1); mean = a1@Wa2p+ba2 (waves 0-1, WMMA)
// c1 = tanh(ys@Wc1+bc1); value = c1@Wc2p+bc2 (waves 2-3, WMMA, N=0 column)
__global__ __launch_bounds__(256) void k_heads(
    const bf16* __restrict__ ys_bf,
    const bf16* __restrict__ Wa1_p, const float* __restrict__ ba1,
    const bf16* __restrict__ Wa2_p, const float* __restrict__ ba2,
    const bf16* __restrict__ Wc1_p, const float* __restrict__ bc1,
    const bf16* __restrict__ Wc2_p, const float* __restrict__ bc2,
    float* __restrict__ mean_out, float* __restrict__ value_out) {
  __shared__ alignas(32) bf16 s_ys[32 * H_];
  __shared__ alignas(32) bf16 s_a1[32 * H_];
  __shared__ alignas(32) bf16 s_c1[32 * H_];
  const int a = blockIdx.x;
  const int m0 = blockIdx.y * 32;
  const int tid = threadIdx.x, lane = tid & 31, wave = tid >> 5;

  const bf16* ys_a = ys_bf + ((size_t)a * M_ + m0) * H_;
  {
    const uint4* s4 = (const uint4*)ys_a;
    uint4* d4 = (uint4*)s_ys;
    d4[tid] = s4[tid];
    d4[tid + 256] = s4[tid + 256];
  }
  __syncthreads();

  const v16bf* Wa1p = (const v16bf*)Wa1_p + (size_t)a * (H_ * H_ / 16);
  const v16bf* Wc1p = (const v16bf*)Wc1_p + (size_t)a * (H_ * H_ / 16);
  v8f aa0 = {}, aa1 = {}, ca0 = {}, ca1 = {};
#pragma unroll
  for (int kk = 0; kk < 4; ++kk) {
    v16bf af0 = load_a_frag(s_ys, lane, 0, kk, H_);
    v16bf af1 = load_a_frag(s_ys, lane, 1, kk, H_);
    v16bf bwa = load_b_packed(Wa1p, lane, wave, kk, 4);
    v16bf bwc = load_b_packed(Wc1p, lane, wave, kk, 4);
    aa0 = wmma_bf16(af0, bwa, aa0);
    aa1 = wmma_bf16(af1, bwa, aa1);
    ca0 = wmma_bf16(af0, bwc, ca0);
    ca1 = wmma_bf16(af1, bwc, ca1);
  }
  {
    const int col = wave * 16 + (lane & 15);
    const float b1 = ba1[a * H_ + col];
    const float b2 = bc1[a * H_ + col];
    const int rb = (lane >> 4) * 8;
#pragma unroll
    for (int v = 0; v < 8; ++v) {
      s_a1[(rb + v) * H_ + col]      = (bf16)FAST_TANH(aa0[v] + b1);
      s_a1[(16 + rb + v) * H_ + col] = (bf16)FAST_TANH(aa1[v] + b1);
      s_c1[(rb + v) * H_ + col]      = (bf16)FAST_TANH(ca0[v] + b2);
      s_c1[(16 + rb + v) * H_ + col] = (bf16)FAST_TANH(ca1[v] + b2);
    }
  }
  __syncthreads();

  // projections via WMMA on padded [128,16] weights:
  //   waves 0-1: mean tiles (r = wave); waves 2-3: value tiles (r = wave-2)
  if (wave < 4) {
    const bool is_mean = (wave < 2);
    const int r = wave & 1;
    const v16bf* Wp = is_mean
        ? (const v16bf*)Wa2_p + (size_t)a * (H_ * 16 / 16)
        : (const v16bf*)Wc2_p + (size_t)a * (H_ * 16 / 16);
    const bf16* srcbuf = is_mean ? s_a1 : s_c1;
    v8f acc = {};
#pragma unroll
    for (int kk = 0; kk < 4; ++kk) {
      v16bf af = load_a_frag(srcbuf, lane, r, kk, H_);
      v16bf bw = load_b_packed(Wp, lane, 0, kk, 4);
      acc = wmma_bf16(af, bw, acc);
    }
    const int col = lane & 15;
    const int rb = r * 16 + (lane >> 4) * 8;
    if (is_mean) {
      if (col < ACT_) {
        const float b2 = ba2[a * ACT_ + col];
#pragma unroll
        for (int v = 0; v < 8; ++v)
          mean_out[((size_t)a * M_ + m0 + rb + v) * ACT_ + col] = acc[v] + b2;
      }
    } else {
      if (col == 0) {
        const float b2 = bc2[a];
#pragma unroll
        for (int v = 0; v < 8; ++v)
          value_out[(size_t)a * M_ + m0 + rb + v] = acc[v] + b2;
      }
    }
  }
}

// ---------------- Workspace layout (bytes) ----------------
static constexpr size_t XI_OFF   = 0;                                          // bf16 [A,M,384]
static constexpr size_t YS_OFF   = XI_OFF   + (size_t)A_ * M_ * G3_ * 2;       // bf16 [A,M,128]
static constexpr size_t WE_OFF   = YS_OFF   + (size_t)A_ * M_ * H_ * 2;        // packed bf16
static constexpr size_t WI_OFF   = WE_OFF   + (size_t)A_ * H_ * H_ * 2;
static constexpr size_t WH_OFF   = WI_OFF   + (size_t)A_ * H_ * G3_ * 2;
static constexpr size_t WA1_OFF  = WH_OFF   + (size_t)A_ * H_ * G3_ * 2;
static constexpr size_t WC1_OFF  = WA1_OFF  + (size_t)A_ * H_ * H_ * 2;
static constexpr size_t WA2P_OFF = WC1_OFF  + (size_t)A_ * H_ * H_ * 2;
static constexpr size_t WC2P_OFF = WA2P_OFF + (size_t)A_ * H_ * 16 * 2;

extern "C" void kernel_launch(void* const* d_in, const int* in_sizes, int n_in,
                              void* d_out, int out_size, void* d_ws, size_t ws_size,
                              hipStream_t stream) {
  (void)in_sizes; (void)n_in; (void)out_size; (void)ws_size;
  const float* hstate  = (const float*)d_in[0];
  const float* obs     = (const float*)d_in[1];
  // d_in[2] = avail_actions (unused, matches reference)
  const float* We      = (const float*)d_in[3];
  const float* be      = (const float*)d_in[4];
  const float* Wi      = (const float*)d_in[5];
  const float* bi      = (const float*)d_in[6];
  const float* Wh      = (const float*)d_in[7];
  const float* bhn     = (const float*)d_in[8];
  const float* Wa1     = (const float*)d_in[9];
  const float* ba1     = (const float*)d_in[10];
  const float* Wa2     = (const float*)d_in[11];
  const float* ba2     = (const float*)d_in[12];
  const float* log_std = (const float*)d_in[13];
  const float* Wc1     = (const float*)d_in[14];
  const float* bc1     = (const float*)d_in[15];
  const float* Wc2     = (const float*)d_in[16];
  const float* bc2     = (const float*)d_in[17];
  const unsigned char* done = (const unsigned char*)d_in[18];

  char* ws = (char*)d_ws;
  bf16* xi     = (bf16*)(ws + XI_OFF);
  bf16* ys     = (bf16*)(ws + YS_OFF);
  bf16* We_p   = (bf16*)(ws + WE_OFF);
  bf16* Wi_p   = (bf16*)(ws + WI_OFF);
  bf16* Wh_p   = (bf16*)(ws + WH_OFF);
  bf16* Wa1_p  = (bf16*)(ws + WA1_OFF);
  bf16* Wc1_p  = (bf16*)(ws + WC1_OFF);
  bf16* Wa2_p  = (bf16*)(ws + WA2P_OFF);
  bf16* Wc2_p  = (bf16*)(ws + WC2P_OFF);

  // Output layout: hT [A,B,H] | mean [A,T,B,ACT] | std [A,ACT] | value [A,T,B]
  float* out   = (float*)d_out;
  float* hT    = out;
  float* mean  = hT + (size_t)A_ * B_ * H_;
  float* stdv  = mean + (size_t)A_ * M_ * ACT_;
  float* value = stdv + (size_t)A_ * ACT_;

  const dim3 blk(256);
  k_pack_b<<<(A_ * H_ * H_  + 255) / 256, blk, 0, stream>>>(We,  We_p,  H_, H_,  H_);
  k_pack_b<<<(A_ * H_ * G3_ + 255) / 256, blk, 0, stream>>>(Wi,  Wi_p,  H_, G3_, G3_);
  k_pack_b<<<(A_ * H_ * G3_ + 255) / 256, blk, 0, stream>>>(Wh,  Wh_p,  H_, G3_, G3_);
  k_pack_b<<<(A_ * H_ * H_  + 255) / 256, blk, 0, stream>>>(Wa1, Wa1_p, H_, H_,  H_);
  k_pack_b<<<(A_ * H_ * H_  + 255) / 256, blk, 0, stream>>>(Wc1, Wc1_p, H_, H_,  H_);
  k_pack_b<<<(A_ * H_ * 16  + 255) / 256, blk, 0, stream>>>(Wa2, Wa2_p, H_, 16,  ACT_);
  k_pack_b<<<(A_ * H_ * 16  + 255) / 256, blk, 0, stream>>>(Wc2, Wc2_p, H_, 16,  1);
  k_std<<<1, 32, 0, stream>>>(log_std, stdv);

  k_emb_xi<<<dim3(A_, M_ / 32), blk, 0, stream>>>(obs, be, bi, We_p, Wi_p, xi);
  k_gru_scan<<<dim3(A_, B_ / 16), blk, 0, stream>>>(hstate, xi, bhn, done, Wh_p, ys, hT);
  k_heads<<<dim3(A_, M_ / 32), blk, 0, stream>>>(ys, Wa1_p, ba1, Wa2_p, ba2,
                                                 Wc1_p, bc1, Wc2_p, bc2, mean, value);
}
